// TopKLoRALinearLegacy_17944373363239
// MI455X (gfx1250) — matrix operand order repaired and checked by reading, not compiled, not run on hardware
//
#include <hip/hip_runtime.h>
#include <hip/hip_bf16.h>
#include <stdint.h>

#define D_IN   2048
#define D_OUT  2048
#define LORA_R 64
#define TOPK   16
#define LORA_SCALE 2.0f   // ALPHA / R = 128 / 64
#define KSTEP  64         // K elements per LDS stage (2 WMMA k-chunks)
#define LDS_STRIDE 72     // bf16 elems; 144B pitch: 16B-aligned, conflict-free

typedef __attribute__((ext_vector_type(16))) __bf16 v16bf;
typedef __attribute__((ext_vector_type(8)))  __bf16 v8bf;
typedef __attribute__((ext_vector_type(4)))  __bf16 v4bf;
typedef __attribute__((ext_vector_type(8)))  float  v8f;
typedef __attribute__((ext_vector_type(4)))  int    v4i;

typedef __attribute__((address_space(1))) v4i gv4i;   // global 16B chunk
typedef __attribute__((address_space(3))) v4i lv4i;   // LDS 16B chunk
typedef __attribute__((address_space(3))) void lds_void;

// ---- async global -> LDS copy, 16B per lane (ASYNCcnt-tracked) ----
static __device__ __forceinline__ void async_cp16(const __bf16* g, __bf16* l) {
#if __has_builtin(__builtin_amdgcn_global_load_async_to_lds_b128)
  __builtin_amdgcn_global_load_async_to_lds_b128((gv4i*)g, (lv4i*)l, 0, 0);
#else
  asm volatile("global_load_async_to_lds_b128 %0, %1, off"
               :: "v"((unsigned)(uintptr_t)(lds_void*)l), "v"(g)
               : "memory");
#endif
}

static __device__ __forceinline__ void wait_async0() {
#if __has_builtin(__builtin_amdgcn_s_wait_asynccnt)
  __builtin_amdgcn_s_wait_asynccnt(0);
#else
  asm volatile("s_wait_asynccnt 0x0" ::: "memory");
#endif
}

// ---- WMMA fragment loaders (CDNA5 16x16x32 bf16 layouts, ISA 7.12.2) ----
// A matrix 16x32: lane = M row; regs hold K {kb..kb+7, kb+16..kb+23}, kb=(lane/16)*8.
static __device__ __forceinline__ v16bf load_frag_a(const __bf16* p, int ld) {
  const int lane = threadIdx.x & 31;
  const int row  = lane & 15;
  const int kb   = (lane >> 4) << 3;            // 0 or 8
  const __bf16* q = p + (size_t)row * ld + kb;
  union { v16bf v; v8bf h[2]; } u;
  u.h[0] = *(const v8bf*)(q);
  u.h[1] = *(const v8bf*)(q + 16);
  return u.v;
}

// B matrix 32x16: lane = N column; regs hold K {kb..kb+7, kb+16..kb+23}, kb=(lane/16)*8.
static __device__ __forceinline__ v16bf load_frag_b(const __bf16* p, int ld) {
  const int lane = threadIdx.x & 31;
  const int col  = lane & 15;
  const int kb   = (lane >> 4) << 3;            // 0 or 8
  const __bf16* q = p + (size_t)col * ld + kb;
  union { v16bf v; v8bf h[2]; } u;
  u.h[0] = *(const v8bf*)(q);
  u.h[1] = *(const v8bf*)(q + 16);
  return u.v;
}

static __device__ __forceinline__ v8f wmma_bf16(v16bf a, v16bf b, v8f c) {
  return __builtin_amdgcn_wmma_f32_16x16x32_bf16(false, a, false, b,
                                                 (short)0, c, false, false);
}

// ---- fused out = x@W^T + bias + z@B^T ----
// 256 threads = 8 waves (2x4); block tile 64x256, LDS stage K=64.
// Async global->LDS double buffering: next stage's fill is in flight during
// the current stage's 2x8 WMMAs; s_wait_asynccnt 0 + barrier per stage.
__global__ __launch_bounds__(256) void fused_gemm_kernel(
    const __bf16* __restrict__ xbf, const __bf16* __restrict__ wbf,
    const __bf16* __restrict__ zbf, const __bf16* __restrict__ bbf,
    const float* __restrict__ bias, float* __restrict__ out) {
  __shared__ __bf16 xs[2][64 * LDS_STRIDE];     // 2 x 9 KB
  __shared__ __bf16 wls[2][256 * LDS_STRIDE];   // 2 x 36 KB

  const int tid  = threadIdx.x;
  const int wave = tid >> 5;
  const int wm   = wave >> 2;                       // 0..1
  const int wn   = wave & 3;                        // 0..3
  const int m0b  = blockIdx.y * 64;
  const int n0b  = blockIdx.x * 256;
  const int m0   = m0b + wm * 32;
  const int n0   = n0b + wn * 64;
  const int lane = tid & 31;
  const int col  = lane & 15;
  const int half = lane >> 4;

  // fill one K-stage (64 rows x + 256 rows W, K=64) -> 10 async ops/thread
  auto fill = [&](int buf, int kk) {
#pragma unroll
    for (int r = 0; r < 2; ++r) {                   // x: 64 rows * 8 segs
      const int ch = tid + r * 256;
      const int row = ch >> 3, seg = ch & 7;
      async_cp16(xbf + (size_t)(m0b + row) * D_IN + kk + seg * 8,
                 &xs[buf][row * LDS_STRIDE + seg * 8]);
    }
#pragma unroll
    for (int r = 0; r < 8; ++r) {                   // W: 256 rows * 8 segs
      const int ch = tid + r * 256;
      const int row = ch >> 3, seg = ch & 7;
      async_cp16(wbf + (size_t)(n0b + row) * D_IN + kk + seg * 8,
                 &wls[buf][row * LDS_STRIDE + seg * 8]);
    }
  };

  v8f acc[2][4] = {};

  fill(0, 0);
  wait_async0();
  __syncthreads();

#pragma unroll 2
  for (int step = 0; step < D_IN / KSTEP; ++step) { // 32 stages
    const int buf = step & 1;
    if (step + 1 < D_IN / KSTEP) fill(buf ^ 1, (step + 1) * KSTEP);

#pragma unroll
    for (int kc = 0; kc < 2; ++kc) {                // 2 k-chunks of 32
      const __bf16* xsb = &xs[buf][(wm * 32) * LDS_STRIDE + kc * 32];
      v16bf a0 = load_frag_a(xsb, LDS_STRIDE);
      v16bf a1 = load_frag_a(xsb + 16 * LDS_STRIDE, LDS_STRIDE);
#pragma unroll
      for (int j = 0; j < 4; ++j) {
        v16bf b = load_frag_b(&wls[buf][(wn * 64 + j * 16) * LDS_STRIDE + kc * 32],
                              LDS_STRIDE);
        acc[0][j] = wmma_bf16(a0, b, acc[0][j]);
        acc[1][j] = wmma_bf16(a1, b, acc[1][j]);
      }
    }
    wait_async0();
    __syncthreads();
  }

  // LoRA up-projection epilogue: K = 64, direct from global (tiny, L2-hot)
#pragma unroll
  for (int kk = 0; kk < LORA_R; kk += 32) {
    v16bf a0 = load_frag_a(zbf + (size_t)m0 * LORA_R + kk, LORA_R);
    v16bf a1 = load_frag_a(zbf + (size_t)(m0 + 16) * LORA_R + kk, LORA_R);
#pragma unroll
    for (int j = 0; j < 4; ++j) {
      v16bf b = load_frag_b(bbf + (size_t)(n0 + j * 16) * LORA_R + kk, LORA_R);
      acc[0][j] = wmma_bf16(a0, b, acc[0][j]);
      acc[1][j] = wmma_bf16(a1, b, acc[1][j]);
    }
  }

  // ---- bias + store (f32) ----
#pragma unroll
  for (int j = 0; j < 4; ++j) {
    const int nc = n0 + j * 16 + col;
    const float bv = bias[nc];
#pragma unroll
    for (int i2 = 0; i2 < 2; ++i2) {
#pragma unroll
      for (int i = 0; i < 8; ++i) {
        const int row = m0 + i2 * 16 + i + half * 8;
        out[(size_t)row * D_OUT + nc] = acc[i2][j][i] + bv;
      }
    }
  }
}

// ---- f32 -> bf16 conversion, 4 elements/thread (pure bandwidth pass) ----
__global__ void cvt_f32_to_bf16(const float* __restrict__ in,
                                __bf16* __restrict__ out, long n4) {
  long i = (long)blockIdx.x * blockDim.x + threadIdx.x;
  const long stride = (long)gridDim.x * blockDim.x;
  for (; i < n4; i += stride) {
    float4 v = ((const float4*)in)[i];
    v4bf r;
    r[0] = (__bf16)v.x; r[1] = (__bf16)v.y;
    r[2] = (__bf16)v.z; r[3] = (__bf16)v.w;
    ((v4bf*)out)[i] = r;
  }
}

// ---- z = x @ A^T, top-16-by-|.| mask, scale, emit bf16 ----
__global__ __launch_bounds__(32) void lora_z_topk_kernel(
    const __bf16* __restrict__ xbf, const __bf16* __restrict__ abf,
    __bf16* __restrict__ zbf) {
  __shared__ float zrow[16 * LORA_R];
  const int m0   = blockIdx.x * 16;
  const int lane = threadIdx.x & 31;
  const int col  = lane & 15;
  const int half = lane >> 4;

  const __bf16* xp = xbf + (size_t)m0 * D_IN;

  v8f c[4] = {};
#pragma unroll
  for (int kk = 0; kk < D_IN; kk += 32) {
    v16bf a = load_frag_a(xp + kk, D_IN);
#pragma unroll
    for (int t = 0; t < 4; ++t) {
      v16bf b = load_frag_b(abf + (size_t)(t * 16) * D_IN + kk, D_IN);
      c[t] = wmma_bf16(a, b, c[t]);
    }
  }
#pragma unroll
  for (int t = 0; t < 4; ++t)
#pragma unroll
    for (int i = 0; i < 8; ++i)
      zrow[(i + half * 8) * LORA_R + t * 16 + col] = c[t][i];
  __syncthreads();

  if (lane < 16) {
    const int row = lane;
    const float* zr = &zrow[row * LORA_R];
    uint64_t used = 0;
    float thresh = 0.f;
    for (int p = 0; p < TOPK; ++p) {
      float best = -1.f; int bi = 0;
      for (int j = 0; j < LORA_R; ++j) {
        if (!((used >> j) & 1ull)) {
          float av = fabsf(zr[j]);
          if (av > best) { best = av; bi = j; }
        }
      }
      used |= (1ull << bi);
      thresh = best;
    }
    __bf16* zo = zbf + (size_t)(m0 + row) * LORA_R;
    for (int j = 0; j < LORA_R; ++j) {
      float v = zr[j];
      float o = (fabsf(v) >= thresh) ? v * LORA_SCALE : 0.f;
      zo[j] = (__bf16)o;
    }
  }
}

extern "C" void kernel_launch(void* const* d_in, const int* in_sizes, int n_in,
                              void* d_out, int out_size, void* d_ws, size_t ws_size,
                              hipStream_t stream) {
  const float* x  = (const float*)d_in[0];   // [4,4096,2048]
  const float* A  = (const float*)d_in[1];   // [64,2048]
  const float* B  = (const float*)d_in[2];   // [2048,64]
  const float* W  = (const float*)d_in[3];   // [2048,2048]
  const float* bb = (const float*)d_in[4];   // [2048]
  float* out = (float*)d_out;
  const long M = (long)in_sizes[0] / D_IN;   // 16384 tokens

  // workspace carve (~75 MB total): xbf, wbf, abf, bbf, zbf
  char* ws = (char*)d_ws;
  size_t off = 0;
  auto carve = [&](size_t bytes) {
    void* p = ws + off;
    off = (off + bytes + 255) & ~(size_t)255;
    return p;
  };
  __bf16* xbf = (__bf16*)carve((size_t)M * D_IN * 2);
  __bf16* wbf = (__bf16*)carve((size_t)D_OUT * D_IN * 2);
  __bf16* abf = (__bf16*)carve((size_t)LORA_R * D_IN * 2);
  __bf16* bbf = (__bf16*)carve((size_t)D_OUT * LORA_R * 2);
  __bf16* zbf = (__bf16*)carve((size_t)M * LORA_R * 2);

  cvt_f32_to_bf16<<<2048, 256, 0, stream>>>(x, xbf, M * (long)D_IN / 4);
  cvt_f32_to_bf16<<<1024, 256, 0, stream>>>(W, wbf, (long)D_OUT * D_IN / 4);
  cvt_f32_to_bf16<<<64,   256, 0, stream>>>(A, abf, (long)LORA_R * D_IN / 4);
  cvt_f32_to_bf16<<<64,   256, 0, stream>>>(B, bbf, (long)D_OUT * LORA_R / 4);

  lora_z_topk_kernel<<<(unsigned)(M / 16), 32, 0, stream>>>(xbf, abf, zbf);

  dim3 grid(D_OUT / 256, (unsigned)(M / 64));
  fused_gemm_kernel<<<grid, 256, 0, stream>>>(xbf, wbf, zbf, bbf, bb, out);
}
